// NearestEmbedEMA_23407571763331
// MI455X (gfx1250) — matrix operand order repaired and checked by reading, not compiled, not run on hardware
//
#include <hip/hip_runtime.h>
#include <hip/hip_bf16.h>
#include <stdint.h>

typedef __attribute__((ext_vector_type(16))) __bf16 v16bf;
typedef __attribute__((ext_vector_type(8)))  __bf16 v8bf;
typedef __attribute__((ext_vector_type(8)))  float  v8f;
typedef __attribute__((ext_vector_type(4)))  int    v4i;

#define BB 32
#define DD 256
#define HH 32
#define WW 32
#define KK 1024
#define NN (BB * HH * WW)   // 32768
#define HW (HH * WW)        // 1024
#define NTILES (NN / 16)    // 2048
#define KTILES (KK / 16)    // 64
#define DTILES (DD / 32)    // 8
#define BTILE_ELEMS (DTILES * 32 * 16)   // 4096 bf16 = 8 KB per B k-tile

union V16U { v16bf v; v8bf h[2]; };

// ---- gfx1250 async global->LDS path (guarded; falls back to direct loads) --
#if __has_builtin(__builtin_amdgcn_global_load_async_to_lds_b128)
#define HAVE_ASYNC_LDS 1
#else
#define HAVE_ASYNC_LDS 0
#endif

#if __has_builtin(__builtin_amdgcn_s_wait_asynccnt)
#define WAIT_ASYNC0() __builtin_amdgcn_s_wait_asynccnt(0)
#else
#define WAIT_ASYNC0() asm volatile("s_wait_asynccnt 0x0" ::: "memory")
#endif

// ---------------------------------------------------------------------------
// Kernel 1: wnorm[k] = sum_d weight[d][k]^2   (f32, K threads)
// ---------------------------------------------------------------------------
__global__ void vq_wnorm_kernel(const float* __restrict__ weight,
                                float* __restrict__ wnorm) {
    int k = blockIdx.x * blockDim.x + threadIdx.x;
    if (k >= KK) return;
    float s = 0.f;
#pragma unroll 4
    for (int d = 0; d < DD; ++d) {
        float v = weight[d * KK + k];
        s += v * v;
    }
    wnorm[k] = s;
}

// ---------------------------------------------------------------------------
// Kernel 2: pack weight (f32, D x K row-major) into bf16 WMMA B-fragments.
// Layout: pbw[((kt*DTILES + dt)*32 + lane)*16 + j]
//   holds weight[dt*32 + 8*(lane/16) + (j&7) + 16*(j/8)][kt*16 + lane%16]
// so each lane's 16 bf16 values (32B) are contiguous -> two b128 loads.
// ---------------------------------------------------------------------------
__global__ void vq_pack_w_kernel(const float* __restrict__ weight,
                                 __bf16* __restrict__ pbw) {
    int tid = blockIdx.x * blockDim.x + threadIdx.x;   // 64*8*32*16 = 262144
    if (tid >= KTILES * BTILE_ELEMS) return;
    int j    = tid & 15;
    int lane = (tid >> 4) & 31;
    int dt   = (tid >> 9) & 7;
    int kt   = tid >> 12;
    int d = dt * 32 + 8 * (lane >> 4) + (j & 7) + 16 * (j >> 3);
    int k = kt * 16 + (lane & 15);
    pbw[tid] = (__bf16)weight[d * KK + k];
}

// ---------------------------------------------------------------------------
// Kernel 3: transpose + convert x (B,D,H,W f32) -> xb (N x D row-major bf16)
// n = b*HW + hw ; writes coalesced (d fastest in output).
// ---------------------------------------------------------------------------
__global__ void vq_xpose_kernel(const float* __restrict__ x,
                                __bf16* __restrict__ xb) {
    int tid = blockIdx.x * blockDim.x + threadIdx.x;   // N*D = 8388608
    if (tid >= NN * DD) return;
    int d  = tid & (DD - 1);
    int n  = tid >> 8;
    int b  = n >> 10;          // n / HW
    int hw = n & (HW - 1);
    xb[tid] = (__bf16)x[(b * DD + d) * HW + hw];
}

// ---------------------------------------------------------------------------
// Kernel 4: fused bf16-WMMA GEMM + argmin.
// One wave per 16-row tile, 4 waves/block. A fragments (16 rows x 256 d)
// held in 64 VGPRs across the whole K loop; 64 k-tiles x 8 WMMAs each.
// B k-tiles (8 KB, shared by all 4 waves) staged into LDS via
// GLOBAL_LOAD_ASYNC_TO_LDS_B128, double buffered (ASYNCcnt + barrier).
// C layout (32-bit 16x16): lane L, vgpr i -> M = i + 8*(L/16), N = L%16.
// ---------------------------------------------------------------------------
__global__ void __launch_bounds__(128)
vq_argmin_kernel(const __bf16* __restrict__ xb,
                 const __bf16* __restrict__ pbw,
                 const float* __restrict__ wnorm,
                 int* __restrict__ amin_i,
                 float* __restrict__ amin_f) {
    const int lane  = threadIdx.x & 31;
    const int wv    = threadIdx.x >> 5;
    const int ntile = blockIdx.x * 4 + wv;
    const int half  = lane >> 4;      // 0: lanes 0-15, 1: lanes 16-31
    const int col   = lane & 15;      // N index this lane owns in C

#if HAVE_ASYNC_LDS
    __shared__ __bf16 bsh[2 * BTILE_ELEMS];   // 2 x 8 KB double buffer
    // cooperative DMA of one 8 KB B k-tile: 512 x 16B chunks over 128 threads
    auto issue_tile = [&](int kt, int buf) {
#pragma unroll
        for (int i = 0; i < 4; ++i) {
            int c = (int)threadIdx.x + 128 * i;        // chunk id
            const __bf16* g = pbw + (size_t)kt * BTILE_ELEMS + c * 8;
            __bf16* l = bsh + buf * BTILE_ELEMS + c * 8;
            __builtin_amdgcn_global_load_async_to_lds_b128(
                (__attribute__((address_space(1))) v4i*)g,
                (__attribute__((address_space(3))) v4i*)l,
                0, 0);
        }
    };
    issue_tile(0, 0);
#endif

    // ---- load all A fragments for this 16-row tile (row = ntile*16 + col) --
    // A layout (16-bit 16x32): lane L (half h) holds K(d) = 8h..8h+7 and
    // 16+8h..16+8h+7 for row M = L%16.
    const __bf16* xrow = xb + (size_t)(ntile * 16 + col) * DD;
    v16bf afrag[DTILES];
#pragma unroll
    for (int dt = 0; dt < DTILES; ++dt) {
        int dbase = dt * 32 + 8 * half;
        V16U u;
        u.h[0] = *(const v8bf*)(xrow + dbase);
        u.h[1] = *(const v8bf*)(xrow + dbase + 16);
        afrag[dt] = u.v;
    }

    float minDist[8];
    int   minIdx[8];
#pragma unroll
    for (int i = 0; i < 8; ++i) { minDist[i] = 3.4e38f; minIdx[i] = 0; }

    // ---- K loop: 64 tiles of 16 codes ------------------------------------
    for (int kt = 0; kt < KTILES; ++kt) {
#if HAVE_ASYNC_LDS
        const int cur = kt & 1;
        WAIT_ASYNC0();          // my chunks of buffer `cur` are in LDS
        __syncthreads();        // everyone's chunks are in LDS
        if (kt + 1 < KTILES) issue_tile(kt + 1, cur ^ 1);  // overlap next DMA
        const __bf16* bbase = bsh + cur * BTILE_ELEMS;
#else
        const __bf16* bbase = pbw + (size_t)kt * BTILE_ELEMS;
        if (kt + 1 < KTILES) {
            __builtin_prefetch(
                (const void*)(pbw + (size_t)(kt + 1) * BTILE_ELEMS + lane * 16),
                0, 3);
        }
#endif
        v8f acc = {};
#pragma unroll
        for (int dt = 0; dt < DTILES; ++dt) {
            const __bf16* bl = bbase + ((size_t)dt * 32 + lane) * 16;
            V16U u;
            u.h[0] = *(const v8bf*)(bl);
            u.h[1] = *(const v8bf*)(bl + 8);
            acc = __builtin_amdgcn_wmma_f32_16x16x32_bf16(
                false, afrag[dt], false, u.v, (short)0, acc, false, false);
        }
        const int   kcol = kt * 16 + col;
        const float wn   = wnorm[kcol];
#pragma unroll
        for (int i = 0; i < 8; ++i) {
            float dist = wn - 2.0f * acc[i];
            if (dist < minDist[i]) { minDist[i] = dist; minIdx[i] = kcol; }
        }
#if HAVE_ASYNC_LDS
        __syncthreads();        // all waves done with buffer `cur`
#endif
    }

    // ---- reduce over the 16 columns held by the 16 lanes of each half ----
#pragma unroll
    for (int i = 0; i < 8; ++i) {
        float d  = minDist[i];
        int   ix = minIdx[i];
#pragma unroll
        for (int off = 8; off >= 1; off >>= 1) {
            float od  = __shfl_xor(d, off, 16);
            int   oix = __shfl_xor(ix, off, 16);
            if (od < d || (od == d && oix < ix)) { d = od; ix = oix; }
        }
        if (col == 0) {
            int row = ntile * 16 + i + 8 * half;   // C-layout row for vgpr i
            amin_i[row] = ix;
            amin_f[row] = (float)ix;
        }
    }
}

// ---------------------------------------------------------------------------
// Kernel 5: gather codes back to (B, D, H, W) layout. Coalesced stores;
// weight row (4 KB) gathers hit L2.
// ---------------------------------------------------------------------------
__global__ void vq_gather_kernel(const float* __restrict__ weight,
                                 const int* __restrict__ amin_i,
                                 float* __restrict__ out) {
    int tid = blockIdx.x * blockDim.x + threadIdx.x;   // B*D*H*W = 8388608
    if (tid >= BB * DD * HW) return;
    int hw = tid & (HW - 1);
    int d  = (tid >> 10) & (DD - 1);
    int b  = tid >> 18;
    int n  = b * HW + hw;
    int k  = amin_i[n];
    out[tid] = weight[d * KK + k];
}

// ---------------------------------------------------------------------------
extern "C" void kernel_launch(void* const* d_in, const int* in_sizes, int n_in,
                              void* d_out, int out_size, void* d_ws, size_t ws_size,
                              hipStream_t stream) {
    const float* x      = (const float*)d_in[0];   // (B, D, H, W)
    const float* weight = (const float*)d_in[1];   // (D, K)

    // d_out: result (B*D*H*W floats) then argmin (N values as float)
    float* out_res  = (float*)d_out;
    float* out_amin = (float*)d_out + (size_t)BB * DD * HW;

    // workspace layout
    char* ws = (char*)d_ws;
    __bf16* xb    = (__bf16*)ws;                               // 16 MB
    __bf16* pbw   = (__bf16*)(ws + (size_t)NN * DD * 2);       // 512 KB
    float*  wnorm = (float*)(ws + (size_t)NN * DD * 2 + (size_t)KK * DD * 2);
    int*    amin  = (int*)((char*)wnorm + KK * sizeof(float));

    vq_wnorm_kernel<<<(KK + 255) / 256, 256, 0, stream>>>(weight, wnorm);

    vq_pack_w_kernel<<<(KTILES * BTILE_ELEMS + 255) / 256, 256, 0, stream>>>(
        weight, pbw);

    vq_xpose_kernel<<<((size_t)NN * DD + 255) / 256, 256, 0, stream>>>(x, xb);

    // 4 waves/block, one 16-row tile per wave -> 512 blocks
    vq_argmin_kernel<<<NTILES / 4, 128, 0, stream>>>(xb, pbw, wnorm, amin, out_amin);

    vq_gather_kernel<<<((size_t)BB * DD * HW + 255) / 256, 256, 0, stream>>>(
        weight, amin, out_res);
}
